// MultiHeadLatentAttention_88046829568482
// MI455X (gfx1250) — matrix operand order repaired and checked by reading, not compiled, not run on hardware
//
#include <hip/hip_runtime.h>

// ---------------------------------------------------------------------------
// MLA forward for MI455X (gfx1250, wave32, WMMA bf16 -> f32 accumulate)
// B=2, T=2048, D=2048, H=16, DH=128, DL=512
// ---------------------------------------------------------------------------

typedef __bf16 bf16;
typedef __attribute__((ext_vector_type(16))) __bf16 bf16x16;
typedef __attribute__((ext_vector_type(8)))  __bf16 bf16x8;
typedef __attribute__((ext_vector_type(4)))  __bf16 bf16x4;
typedef __attribute__((ext_vector_type(8)))  float  f32x8;

static __device__ __forceinline__ bf16x16 cat8(bf16x8 lo, bf16x8 hi) {
  return __builtin_shufflevector(lo, hi, 0,1,2,3,4,5,6,7,8,9,10,11,12,13,14,15);
}

// A-matrix fragment (16x32 bf16, K contiguous in memory):
// lanes 0-15 hold K {0..7, 16..23}; lanes 16-31 hold K {8..15, 24..31}
static __device__ __forceinline__ bf16x16 load_a_frag(const bf16* p, int half) {
  const int ka = half ? 8 : 0;
  bf16x8 lo = *(const bf16x8*)(p + ka);
  bf16x8 hi = *(const bf16x8*)(p + 16 + ka);
  return cat8(lo, hi);
}

// B-matrix fragment (32x16 bf16, stored K-contiguous per column, lane = column):
// lanes 0-15 hold K {0..7, 16..23}; lanes 16-31 hold K {8..15, 24..31}
static __device__ __forceinline__ bf16x16 load_b_frag(const bf16* p, int half) {
  const int kb = half ? 8 : 0;
  bf16x8 lo = *(const bf16x8*)(p + kb);
  bf16x8 hi = *(const bf16x8*)(p + kb + 16);
  return cat8(lo, hi);
}

static __device__ __forceinline__ f32x8 wmma_bf16(bf16x16 a, bf16x16 b, f32x8 c) {
  return __builtin_amdgcn_wmma_f32_16x16x32_bf16(false, a, false, b, (short)0, c,
                                                 false, false);
}

// ---------------------------------------------------------------------------
// Elementwise f32 -> bf16 (vectorized x4)
// ---------------------------------------------------------------------------
__global__ __launch_bounds__(256)
void f32_to_bf16_v4(const float4* __restrict__ in, bf16* __restrict__ out, int n4) {
  int i = blockIdx.x * 256 + threadIdx.x;
  if (i < n4) {
    float4 v = in[i];
    bf16x4 o;
    o.x = (bf16)v.x; o.y = (bf16)v.y; o.z = (bf16)v.z; o.w = (bf16)v.w;
    *(bf16x4*)(out + 4 * (size_t)i) = o;
  }
}

// ---------------------------------------------------------------------------
// Weight transpose-convert: W f32 [K,N] -> WT bf16 [N,K] row-major.
// 32x32 LDS tile so both global read and write are coalesced.
// Block: 32x8 threads; each thread handles 4 rows of the tile.
// ---------------------------------------------------------------------------
__global__ __launch_bounds__(256)
void transpose_w_bf16(const float* __restrict__ W, bf16* __restrict__ WT,
                      int K, int N) {
  __shared__ float tile[32][33];
  const int tx = threadIdx.x & 31;
  const int ty = threadIdx.x >> 5;          // 0..7
  const int nb = blockIdx.x * 32;           // n tile base
  const int kb = blockIdx.y * 32;           // k tile base
#pragma unroll
  for (int j = 0; j < 32; j += 8)
    tile[ty + j][tx] = W[(size_t)(kb + ty + j) * N + nb + tx];
  __syncthreads();
#pragma unroll
  for (int j = 0; j < 32; j += 8)
    WT[(size_t)(nb + ty + j) * K + kb + tx] = (bf16)tile[tx][ty + j];
}

// ---------------------------------------------------------------------------
// bf16 GEMM: C[M,N] = A[M,K] @ BT[N,K]^T + bias[N]
// 256 threads = 8 waves; block tile 64(M) x 256(N); wave tile 32(M) x 64(N).
// Each B fragment feeds two WMMAs (two A row-fragments).
// MODE 0: store bf16 [M,N]; 1: store f32 [M,N]; 2: store bf16 transposed
//         per head as [B, H, DH, T]  (for V)
// ---------------------------------------------------------------------------
template <int MODE>
__global__ __launch_bounds__(256)
void gemm_bf16_wmma(const bf16* __restrict__ A, const bf16* __restrict__ BT,
                    const float* __restrict__ bias, void* __restrict__ Cout,
                    int M, int N, int K) {
  const int lane = threadIdx.x & 31;
  const int wid  = threadIdx.x >> 5;
  const int half = lane >> 4;
  const int c16  = lane & 15;

  const int mtiles = M >> 6;
  const int mblk = blockIdx.x % mtiles;
  const int nblk = blockIdx.x / mtiles;
  const int m0 = mblk * 64 + (wid & 1) * 32;     // 2 M-waves x 32 rows
  const int n0 = nblk * 256 + (wid >> 1) * 64;   // 4 N-waves x 64 cols

  f32x8 acc[2][4] = {};
  const bf16* arow0 = A + (size_t)(m0 + c16) * K;
  const bf16* arow1 = A + (size_t)(m0 + 16 + c16) * K;

  for (int kk = 0; kk < K; kk += 32) {
    if (kk + 256 < K) {
      __builtin_prefetch(arow0 + kk + 256, 0, 1);
      __builtin_prefetch(arow1 + kk + 256, 0, 1);
    }
    bf16x16 af0 = load_a_frag(arow0 + kk, half);
    bf16x16 af1 = load_a_frag(arow1 + kk, half);
#pragma unroll
    for (int nt = 0; nt < 4; ++nt) {
      const bf16* bcol = BT + (size_t)(n0 + nt * 16 + c16) * K + kk;
      bf16x16 bf = load_b_frag(bcol, half);
      acc[0][nt] = wmma_bf16(af0, bf, acc[0][nt]);
      acc[1][nt] = wmma_bf16(af1, bf, acc[1][nt]);
    }
  }

#pragma unroll
  for (int mi = 0; mi < 2; ++mi) {
#pragma unroll
    for (int nt = 0; nt < 4; ++nt) {
      const int n = n0 + nt * 16 + c16;
      const float bv = bias[n];
#pragma unroll
      for (int i = 0; i < 8; ++i) {
        const int m = m0 + mi * 16 + i + half * 8;   // C layout
        const float val = acc[mi][nt][i] + bv;
        if (MODE == 1) {
          ((float*)Cout)[(size_t)m * N + n] = val;
        } else if (MODE == 0) {
          ((bf16*)Cout)[(size_t)m * N + n] = (bf16)val;
        } else {
          // [B, H, DH, T] with T=2048 (m = b*2048+t), N column = h*128+d
          const int b = m >> 11, t = m & 2047;
          const int h = n >> 7,  d = n & 127;
          ((bf16*)Cout)[((((size_t)b * 16 + h) * 128 + d) << 11) + t] = (bf16)val;
        }
      }
    }
  }
}

// ---------------------------------------------------------------------------
// Flash attention: one wave per (b, h, 32-query tile). Causal.
// Each K/V fragment load feeds two WMMAs (two 16-row q sub-tiles).
// Q  : bf16 [B*T, D]   (head h at column h*128)
// Kf : bf16 [B*T, D]
// VT : bf16 [B, H, DH, T]   (transposed V)
// Mid: bf16 [B*T, D]
// ---------------------------------------------------------------------------
__global__ __launch_bounds__(256)
void mla_attention(const bf16* __restrict__ Q, const bf16* __restrict__ Kf,
                   const bf16* __restrict__ VT, bf16* __restrict__ Mid) {
  constexpr int Tn = 2048, Dc = 2048, Hc = 16, DH = 128;
  __shared__ bf16 ptile[8][32 * 32];

  const int lane = threadIdx.x & 31;
  const int wid  = threadIdx.x >> 5;
  const int half = lane >> 4;
  const int c16  = lane & 15;

  const int task = blockIdx.x * 8 + wid;     // 0 .. 2047
  const int qt = task & 63;                  // 64 q-tiles of 32 rows
  const int h  = (task >> 6) & 15;
  const int b  = task >> 10;
  const int q0 = qt * 32;

  // Q fragments: 2 sub-tiles x 4 dh-chunks
  bf16x16 qf[2][4];
#pragma unroll
  for (int qr = 0; qr < 2; ++qr) {
    const bf16* qrow = Q + ((size_t)b * Tn + q0 + qr * 16 + c16) * Dc + h * DH;
#pragma unroll
    for (int c = 0; c < 4; ++c) qf[qr][c] = load_a_frag(qrow + c * 32, half);
  }

  f32x8 o[2][8] = {};
  float mrow[2][8], lrow[2][8];
#pragma unroll
  for (int qr = 0; qr < 2; ++qr)
#pragma unroll
    for (int i = 0; i < 8; ++i) { mrow[qr][i] = -1e30f; lrow[qr][i] = 0.0f; }

  const float scale = 0.088388347648318447f;  // 1/sqrt(128)
  const bf16* kbase = Kf + (size_t)b * Tn * Dc + h * DH;
  const bf16* vbase = VT + ((size_t)b * Hc + h) * DH * (size_t)Tn;
  bf16* ldsP = ptile[wid];

  const int nk = (q0 + 32 + 31) >> 5;        // causal: keys < q0+32
  for (int kt = 0; kt < nk; ++kt) {
    const int k0 = kt * 32;

    // scores: 2 q sub-tiles x 2 key col-tiles
    f32x8 s[2][2] = {};
#pragma unroll
    for (int c = 0; c < 4; ++c) {
      bf16x16 kf0 = load_b_frag(kbase + (size_t)(k0 + c16) * Dc + c * 32, half);
      s[0][0] = wmma_bf16(qf[0][c], kf0, s[0][0]);
      s[1][0] = wmma_bf16(qf[1][c], kf0, s[1][0]);
      bf16x16 kf1 = load_b_frag(kbase + (size_t)(k0 + 16 + c16) * Dc + c * 32, half);
      s[0][1] = wmma_bf16(qf[0][c], kf1, s[0][1]);
      s[1][1] = wmma_bf16(qf[1][c], kf1, s[1][1]);
    }

    // online softmax (row = qr*16 + i + 8*half, cols = c16 per half)
#pragma unroll
    for (int qr = 0; qr < 2; ++qr) {
#pragma unroll
      for (int i = 0; i < 8; ++i) {
        const int qi = q0 + qr * 16 + i + half * 8;
        float v0 = s[qr][0][i] * scale;
        float v1 = s[qr][1][i] * scale;
        if (k0 + c16 > qi)      v0 = -1e30f;
        if (k0 + 16 + c16 > qi) v1 = -1e30f;
        float rm = fmaxf(v0, v1);
#pragma unroll
        for (int d = 1; d < 16; d <<= 1) rm = fmaxf(rm, __shfl_xor(rm, d, 16));
        const float mnew = fmaxf(mrow[qr][i], rm);
        const float cf = __expf(mrow[qr][i] - mnew);
        mrow[qr][i] = mnew;
        const float p0 = __expf(v0 - mnew);
        const float p1 = __expf(v1 - mnew);
        float ls = p0 + p1;
#pragma unroll
        for (int d = 1; d < 16; d <<= 1) ls += __shfl_xor(ls, d, 16);
        lrow[qr][i] = lrow[qr][i] * cf + ls;
        // rescale running output row immediately
#pragma unroll
        for (int dt = 0; dt < 8; ++dt) o[qr][dt][i] *= cf;
        const int r = qr * 16 + i + half * 8;
        ldsP[r * 32 + c16]      = (bf16)p0;
        ldsP[r * 32 + 16 + c16] = (bf16)p1;
      }
    }

    // C-layout -> A-layout through LDS (same wave; order DS ops)
    asm volatile("s_wait_dscnt 0" ::: "memory");
    bf16x16 pf0 = load_a_frag(ldsP + (c16) * 32, half);
    bf16x16 pf1 = load_a_frag(ldsP + (16 + c16) * 32, half);

    // O += P @ V : each V fragment feeds both q sub-tiles
#pragma unroll
    for (int dt = 0; dt < 8; ++dt) {
      const bf16* vr = vbase + (size_t)(dt * 16 + c16) * Tn + k0;
      bf16x16 vf = load_b_frag(vr, half);
      o[0][dt] = wmma_bf16(pf0, vf, o[0][dt]);
      o[1][dt] = wmma_bf16(pf1, vf, o[1][dt]);
    }
    asm volatile("s_wait_dscnt 0" ::: "memory");
  }

  // normalize + store bf16 [B*T, D]
#pragma unroll
  for (int qr = 0; qr < 2; ++qr) {
#pragma unroll
    for (int i = 0; i < 8; ++i) {
      const float inv = 1.0f / lrow[qr][i];
      bf16* orow =
          Mid + ((size_t)b * Tn + q0 + qr * 16 + i + half * 8) * Dc + h * DH;
#pragma unroll
      for (int dt = 0; dt < 8; ++dt)
        orow[dt * 16 + c16] = (bf16)(o[qr][dt][i] * inv);
    }
  }
}

// ---------------------------------------------------------------------------
// Host-side launch
// ---------------------------------------------------------------------------
extern "C" void kernel_launch(void* const* d_in, const int* in_sizes, int n_in,
                              void* d_out, int out_size, void* d_ws, size_t ws_size,
                              hipStream_t stream) {
  (void)in_sizes; (void)n_in; (void)out_size; (void)ws_size;
  const float* hidden = (const float*)d_in[0];
  // d_in[1] = causal_mask (computed analytically instead)
  const float* Wq  = (const float*)d_in[2];
  const float* bq  = (const float*)d_in[3];
  const float* Wkl = (const float*)d_in[4];
  const float* bkl = (const float*)d_in[5];
  const float* Wvl = (const float*)d_in[6];
  const float* bvl = (const float*)d_in[7];
  const float* Wke = (const float*)d_in[8];
  const float* bke = (const float*)d_in[9];
  const float* Wve = (const float*)d_in[10];
  const float* bve = (const float*)d_in[11];
  const float* Wo  = (const float*)d_in[12];
  const float* bo  = (const float*)d_in[13];

  constexpr int Bc = 2, Tc = 2048, Dc = 2048, DLc = 512;
  constexpr int Mc = Bc * Tc;  // 4096

  char* ws = (char*)d_ws;
  size_t off = 0;
  auto alloc = [&](size_t bytes) -> void* {
    void* p = ws + off;
    off += (bytes + 255) & ~(size_t)255;
    return p;
  };
  bf16* Xb   = (bf16*)alloc((size_t)Mc * Dc * 2);
  bf16* WqT  = (bf16*)alloc((size_t)Dc * Dc * 2);
  bf16* WklT = (bf16*)alloc((size_t)DLc * Dc * 2);
  bf16* WvlT = (bf16*)alloc((size_t)DLc * Dc * 2);
  bf16* WkeT = (bf16*)alloc((size_t)Dc * DLc * 2);
  bf16* WveT = (bf16*)alloc((size_t)Dc * DLc * 2);
  bf16* WoT  = (bf16*)alloc((size_t)Dc * Dc * 2);
  bf16* Qb   = (bf16*)alloc((size_t)Mc * Dc * 2);
  bf16* KLb  = (bf16*)alloc((size_t)Mc * DLc * 2);
  bf16* VLb  = (bf16*)alloc((size_t)Mc * DLc * 2);
  bf16* Kfb  = (bf16*)alloc((size_t)Mc * Dc * 2);
  bf16* VTb  = (bf16*)alloc((size_t)Mc * Dc * 2);
  bf16* Mid  = (bf16*)alloc((size_t)Mc * Dc * 2);

  // --- conversions ---
  {
    const int n4 = Mc * Dc / 4;
    f32_to_bf16_v4<<<(n4 + 255) / 256, 256, 0, stream>>>((const float4*)hidden, Xb, n4);
  }
  auto tr = [&](const float* W, bf16* WT, int K, int N) {
    transpose_w_bf16<<<dim3(N / 32, K / 32), 256, 0, stream>>>(W, WT, K, N);
  };
  tr(Wq,  WqT,  Dc,  Dc);
  tr(Wkl, WklT, Dc,  DLc);
  tr(Wvl, WvlT, Dc,  DLc);
  tr(Wke, WkeT, DLc, Dc);
  tr(Wve, WveT, DLc, Dc);
  tr(Wo,  WoT,  Dc,  Dc);

  // --- projections ---
  auto gemm_grid = [](int M, int N) { return dim3((M / 64) * (N / 256)); };
  gemm_bf16_wmma<0><<<gemm_grid(Mc, Dc),  256, 0, stream>>>(Xb,  WqT,  bq,  Qb,  Mc, Dc,  Dc);
  gemm_bf16_wmma<0><<<gemm_grid(Mc, DLc), 256, 0, stream>>>(Xb,  WklT, bkl, KLb, Mc, DLc, Dc);
  gemm_bf16_wmma<0><<<gemm_grid(Mc, DLc), 256, 0, stream>>>(Xb,  WvlT, bvl, VLb, Mc, DLc, Dc);
  gemm_bf16_wmma<0><<<gemm_grid(Mc, Dc),  256, 0, stream>>>(KLb, WkeT, bke, Kfb, Mc, Dc,  DLc);
  gemm_bf16_wmma<2><<<gemm_grid(Mc, Dc),  256, 0, stream>>>(VLb, WveT, bve, VTb, Mc, Dc,  DLc);

  // --- attention: 2048 wave-tasks, 8 waves/block ---
  mla_attention<<<256, 256, 0, stream>>>(Qb, Kfb, VTb, Mid);

  // --- output projection (f32 out) ---
  gemm_bf16_wmma<1><<<gemm_grid(Mc, Dc),  256, 0, stream>>>(Mid, WoT, bo, (float*)d_out, Mc, Dc, Dc);
}
